// CausalSelfAttention_89197880803485
// MI455X (gfx1250) — matrix operand order repaired and checked by reading, not compiled, not run on hardware
//
#include <hip/hip_runtime.h>

// ---------------------------------------------------------------------------
// CDNA5 / gfx1250 causal self-attention forward (bf16 WMMA, fp32 accumulate)
// B=2, T=2048, DIM=1024, HEADS=16, HEAD_DIM=64
//
// - GEMMs: double-buffered LDS fed by global_load_async_to_lds_b128 (ASYNCcnt)
// - Attention: K/V^T tiles staged block-wide in LDS by the Tensor Data Mover
//   (TENSOR_LOAD_TO_LDS, TENSORcnt), double-buffered against the WMMAs.
// ---------------------------------------------------------------------------

typedef __bf16 bf16;
typedef __attribute__((ext_vector_type(8)))  __bf16 v8bf;
typedef __attribute__((ext_vector_type(16))) __bf16 v16bf;
typedef __attribute__((ext_vector_type(8)))  float  v8f;
typedef __attribute__((ext_vector_type(4)))  unsigned int u32x4;
typedef __attribute__((ext_vector_type(8)))  int i32x8;
typedef __attribute__((ext_vector_type(4)))  int i32x4;

#define B_     2
#define T_     2048
#define DIM_   1024
#define H_     16
#define HD_    64
#define INNER_ 1024          // H_*HD_
#define MROWS_ 4096          // B_*T_

// ---- WMMA helpers ---------------------------------------------------------

__device__ __forceinline__ v8f wmma_bf16(v16bf a, v16bf b, v8f c) {
  return __builtin_amdgcn_wmma_f32_16x16x32_bf16(
      false, a, false, b, (short)0, c, false, false);
}

// A-matrix 16x32 bf16 fragment: lane<16 holds M=lane, K=0..7 then 16..23;
// lane>=16 holds K=8..15 then 24..31. Source row-major [16][stride].
__device__ __forceinline__ v16bf frag_a(const bf16* base, int stride, int lane) {
  int r = lane & 15, h = lane >> 4;
  const bf16* p = base + r * stride + h * 8;
  v8bf lo = *(const v8bf*)(p);
  v8bf hi = *(const v8bf*)(p + 16);
  return __builtin_shufflevector(lo, hi, 0,1,2,3,4,5,6,7,8,9,10,11,12,13,14,15);
}

// B-matrix 32x16 bf16 fragment: lane<16 holds N=lane, K=0..7 then 16..23;
// lane>=16 holds K=8..15 then 24..31. Source is [N][K] row-major (B transposed).
__device__ __forceinline__ v16bf frag_b(const bf16* base, int stride, int lane) {
  int n = lane & 15, h = lane >> 4;
  const bf16* p = base + n * stride + h * 8;
  v8bf lo = *(const v8bf*)(p);
  v8bf hi = *(const v8bf*)(p + 16);
  return __builtin_shufflevector(lo, hi, 0,1,2,3,4,5,6,7,8,9,10,11,12,13,14,15);
}

__device__ __forceinline__ float hmax16(float v) {
  v = fmaxf(v, __shfl_xor(v, 1));
  v = fmaxf(v, __shfl_xor(v, 2));
  v = fmaxf(v, __shfl_xor(v, 4));
  v = fmaxf(v, __shfl_xor(v, 8));
  return v;
}
__device__ __forceinline__ float hsum16(float v) {
  v += __shfl_xor(v, 1);
  v += __shfl_xor(v, 2);
  v += __shfl_xor(v, 4);
  v += __shfl_xor(v, 8);
  return v;
}

// ---- gfx1250 async copy (memory -> LDS, ASYNCcnt) -------------------------

__device__ __forceinline__ unsigned lds_off(const void* p) {
  // generic LDS address = {shared aperture, byte offset}; low 32 bits = offset
  return (unsigned)(unsigned long long)p;
}

__device__ __forceinline__ void async_b128(unsigned lds, const bf16* g) {
  asm volatile("global_load_async_to_lds_b128 %0, %1, off"
               :: "v"(lds), "v"((unsigned long long)g) : "memory");
}

__device__ __forceinline__ void wait_async0() {
  asm volatile("s_wait_asynccnt 0x0" ::: "memory");
}

// ---- gfx1250 Tensor Data Mover: 2-D tile, global -> LDS (TENSORcnt) -------
// D# per CDNA5 ISA 8.3/8.4:
//  group0: [1:0]=count(1) | lds_addr[63:32] | global_addr[120:64] | type=2
//  group1: data_size[17:16]=1 (2B) | tensor_dim0[79:48] | tensor_dim1[111:80]
//          | tile_dim0[127:112] | tile_dim1[143:128] | dim0_stride[207:160]
//  groups 2/3: zero (tensor <= 2D)

__device__ __forceinline__ void tdm_load_2d(unsigned lds_addr, const void* gptr,
                                            unsigned tile_d0, unsigned tile_d1,
                                            unsigned tensor_d0, unsigned tensor_d1,
                                            unsigned long long stride0) {
  unsigned long long ga = (unsigned long long)gptr;
  u32x4 g0;
  g0[0] = 1u;                                            // count=1 (user D#)
  g0[1] = lds_addr;                                      // LDS byte address
  g0[2] = (unsigned)(ga & 0xffffffffu);                  // global_addr[31:0]
  g0[3] = (unsigned)((ga >> 32) & 0x01ffffffu) | (2u << 30);  // [56:32]|type=2
  i32x8 g1;
  g1[0] = (int)(1u << 16);                               // data_size=1 -> 2B
  g1[1] = (int)((tensor_d0 & 0xffffu) << 16);            // tensor_dim0 lo
  g1[2] = (int)((tensor_d0 >> 16) | ((tensor_d1 & 0xffffu) << 16));
  g1[3] = (int)((tensor_d1 >> 16) | (tile_d0 << 16));    // tile_dim0
  g1[4] = (int)(tile_d1 & 0xffffu);                      // tile_dim1, dim2=0
  g1[5] = (int)(unsigned)(stride0 & 0xffffffffu);        // dim0_stride lo
  g1[6] = (int)(unsigned)((stride0 >> 32) & 0xffffu);    // stride hi, d1s=0
  g1[7] = 0;
  i32x4 z4 = {0, 0, 0, 0};
  i32x8 z8 = {0, 0, 0, 0, 0, 0, 0, 0};
  // 6-arg form on this toolchain: (g0, g1, g2, g3, <extra group>, cpol)
  __builtin_amdgcn_tensor_load_to_lds(g0, g1, z4, z4, z8, 0);
}

// ---- fp32 -> bf16 conversion ----------------------------------------------

__global__ __launch_bounds__(256)
void cvt_kernel(const float* __restrict__ in, bf16* __restrict__ out, long n) {
  long i = ((long)blockIdx.x * 256 + threadIdx.x) * 4;
  if (i + 3 < n) {
    float4 f = *(const float4*)(in + i);
    out[i + 0] = (bf16)f.x;
    out[i + 1] = (bf16)f.y;
    out[i + 2] = (bf16)f.z;
    out[i + 3] = (bf16)f.w;
  }
}

// in[K][N] fp32 -> out[N][K] bf16 (weights transposed to B-fragment layout)
__global__ __launch_bounds__(256)
void cvt_t_kernel(const float* __restrict__ in, bf16* __restrict__ out,
                  int K, int N) {
  long i = (long)blockIdx.x * 256 + threadIdx.x;
  if (i < (long)K * N) {
    int nn = (int)(i % N);
    int kk = (int)(i / N);
    out[(size_t)nn * K + kk] = (bf16)in[i];
  }
}

// ---- WMMA GEMM: C[M][N] f32 = A[M][K] bf16 @ Bt[N][K] bf16 ----------------
// Block: 256 threads = 8 waves; block tile 128(M) x 256(N), K-step 32.
// Wave grid 2(M) x 4(N): each wave computes 64x64 = 4x4 WMMA tiles.
// LDS double-buffered, filled by global_load_async_to_lds_b128.

#define BM_ 128
#define BN_ 256
#define BK_ 32

__global__ __launch_bounds__(256)
void gemm_bf16_kernel(const bf16* __restrict__ A, const bf16* __restrict__ Bt,
                      float* __restrict__ C, int M, int N, int K) {
  __shared__ bf16 sA[2][BM_][BK_];   // 2 x 8 KB
  __shared__ bf16 sB[2][BN_][BK_];   // 2 x 16 KB
  int tid  = threadIdx.x;
  int wave = tid >> 5, lane = tid & 31;
  int bm = blockIdx.y * BM_, bn = blockIdx.x * BN_;
  int wm = (wave & 1) * 64;          // 2 waves across M
  int wn = (wave >> 1) * 64;         // 4 waves across N

  v8f acc[4][4];
#pragma unroll
  for (int i = 0; i < 4; ++i)
#pragma unroll
    for (int j = 0; j < 4; ++j)
      acc[i][j] = (v8f){0.f,0.f,0.f,0.f,0.f,0.f,0.f,0.f};

  // Async loader mapping:
  //   A tile 128x32: 2 threads/row, each thread 32B (2 x b128)
  //   B tile 256x32: 1 thread/row, each thread 64B (4 x b128)
  int arow = tid >> 1;
  int acol = (tid & 1) * 16;         // element offset (bf16)
  const bf16* gA = A  + (size_t)(bm + arow) * K + acol;
  const bf16* gB = Bt + (size_t)(bn + tid) * K;

  // prologue: DMA tile 0 into buffer 0
  {
    unsigned la = lds_off(&sA[0][arow][acol]);
    async_b128(la,      gA);
    async_b128(la + 16, gA + 8);
    unsigned lb = lds_off(&sB[0][tid][0]);
    async_b128(lb,      gB);
    async_b128(lb + 16, gB + 8);
    async_b128(lb + 32, gB + 16);
    async_b128(lb + 48, gB + 24);
  }
  wait_async0();
  __syncthreads();

  int nt = K / BK_;
  for (int kt = 0; kt < nt; ++kt) {
    int cur = kt & 1, nxt = cur ^ 1;

    // kick off DMA for tile kt+1 into the other buffer (overlaps WMMAs)
    if (kt + 1 < nt) {
      const bf16* pa = gA + (kt + 1) * BK_;
      unsigned la = lds_off(&sA[nxt][arow][acol]);
      async_b128(la,      pa);
      async_b128(la + 16, pa + 8);
      const bf16* pb = gB + (kt + 1) * BK_;
      unsigned lb = lds_off(&sB[nxt][tid][0]);
      async_b128(lb,      pb);
      async_b128(lb + 16, pb + 8);
      async_b128(lb + 32, pb + 16);
      async_b128(lb + 48, pb + 24);
    }

    // compute on the current buffer
    v16bf af[4], bfr[4];
#pragma unroll
    for (int i = 0; i < 4; ++i)
      af[i] = frag_a(&sA[cur][wm + i * 16][0], BK_, lane);
#pragma unroll
    for (int j = 0; j < 4; ++j)
      bfr[j] = frag_b(&sB[cur][wn + j * 16][0], BK_, lane);
#pragma unroll
    for (int i = 0; i < 4; ++i)
#pragma unroll
      for (int j = 0; j < 4; ++j)
        acc[i][j] = wmma_bf16(af[i], bfr[j], acc[i][j]);

    wait_async0();       // our DMA for kt+1 landed
    __syncthreads();     // everyone finished compute(kt) + DMA(kt+1)
  }

  // C/D layout: VGPR e -> row e (+8 for lanes>=16), col = lane&15.
  int rbase = (lane >> 4) * 8, n = lane & 15;
#pragma unroll
  for (int i = 0; i < 4; ++i) {
#pragma unroll
    for (int j = 0; j < 4; ++j) {
      size_t row0 = (size_t)(bm + wm + i * 16 + rbase);
      size_t col  = (size_t)(bn + wn + j * 16 + n);
#pragma unroll
      for (int e = 0; e < 8; ++e)
        C[(row0 + e) * N + col] = acc[i][j][e];
    }
  }
}

// ---- RoPE + head-major scatter --------------------------------------------
// qkv[MROWS][3*INNER] f32 -> Qh,Kh [B,H,T,64] bf16; Vt [B,H,64,T] bf16.

__global__ __launch_bounds__(256)
void rope_kernel(const float* __restrict__ qkv, bf16* __restrict__ Qh,
                 bf16* __restrict__ Kh, bf16* __restrict__ Vt) {
  size_t idx = (size_t)blockIdx.x * 256 + threadIdx.x;   // over B*T*H*64
  int d = (int)(idx & 63);
  size_t r = idx >> 6;          // bt*H + h
  int h = (int)(r & 15);
  size_t bt = r >> 4;           // b*T + t
  int t = (int)(bt & (T_ - 1));
  int b = (int)(bt >> 11);

  const float* row = qkv + bt * (size_t)(3 * INNER_);
  int base = h * HD_;
  float qv = row[base + d];
  float kv = row[INNER_ + base + d];
  float vv = row[2 * INNER_ + base + d];
  int prt = (d < 32) ? d + 32 : d - 32;
  float qp = row[base + prt];
  float kp = row[INNER_ + base + prt];
  float sgn = (d < 32) ? -1.f : 1.f;

  int i = d & 31;
  // inv_freq = 10000^(-2i/64) = exp(-i * ln(10000)/32)
  float inv_freq = __expf(-0.28782313662425572f * (float)i);
  float ang = (float)t * inv_freq;
  float c = cosf(ang), s = sinf(ang);
  float qr = qv * c + sgn * qp * s;
  float kr = kv * c + sgn * kp * s;

  size_t bh = (size_t)b * H_ + h;
  Qh[(bh * T_ + t) * HD_ + d] = (bf16)qr;
  Kh[(bh * T_ + t) * HD_ + d] = (bf16)kr;
  Vt[(bh * HD_ + d) * T_ + t] = (bf16)vv;
}

// ---- Flash attention (causal, online softmax), all-WMMA --------------------
// grid = (B*H, T/128); 8 waves/block, each wave owns a 16-row q tile.
// K/V^T tiles are staged block-wide in LDS by the Tensor Data Mover
// (wave 0 issues TENSOR_LOAD_TO_LDS, double-buffered; barrier publishes).
// All waves run the block-uniform key loop; causal masking handles the tail.

__global__ __launch_bounds__(256)
void attn_kernel(const bf16* __restrict__ Qh, const bf16* __restrict__ Kh,
                 const bf16* __restrict__ Vt, bf16* __restrict__ Ao) {
  __shared__ bf16 sK[2][32][64];   // [key][d]   2 x 4 KB
  __shared__ bf16 sV[2][64][32];   // [d][key]   2 x 4 KB
  __shared__ bf16 sP[8][16][32];   // per-wave P staging, 8 KB
  int bh = blockIdx.x;
  int b = bh >> 4, h = bh & 15;
  int wave = threadIdx.x >> 5, lane = threadIdx.x & 31;
  int q0 = blockIdx.y * 128 + wave * 16;

  const bf16* Q = Qh + (size_t)bh * T_ * HD_;
  const bf16* K = Kh + (size_t)bh * T_ * HD_;
  const bf16* V = Vt + (size_t)bh * HD_ * T_;   // [d][t]

  int rbase = (lane >> 4) * 8, n = lane & 15;

  v16bf qf0 = frag_a(Q + (size_t)q0 * HD_,      HD_, lane);
  v16bf qf1 = frag_a(Q + (size_t)q0 * HD_ + 32, HD_, lane);

  v8f o0 = {0.f,0.f,0.f,0.f,0.f,0.f,0.f,0.f};
  v8f o1 = o0, o2 = o0, o3 = o0;
  float m[8], l[8];
#pragma unroll
  for (int j = 0; j < 8; ++j) { m[j] = -1e30f; l[j] = 0.f; }

  const float scale = 0.125f;     // 1/sqrt(64)
  int nchunk = (blockIdx.y * 128 + 128) / 32;   // block-uniform causal extent

  // prologue: TDM chunk 0 into buffer 0 (wave 0 drives the DMA)
  if (wave == 0) {
    tdm_load_2d(lds_off(&sK[0][0][0]), K,  HD_, 32, HD_, T_, HD_);
    tdm_load_2d(lds_off(&sV[0][0][0]), V,  32, HD_, T_, HD_, T_);
    __builtin_amdgcn_s_wait_tensorcnt(0);
  }
  __syncthreads();

  for (int c = 0; c < nchunk; ++c) {
    int kc = c * 32;
    int cur = c & 1, nxt = cur ^ 1;

    // TDM for chunk c+1 overlaps this chunk's WMMAs
    if (wave == 0 && c + 1 < nchunk) {
      int kn = kc + 32;
      tdm_load_2d(lds_off(&sK[nxt][0][0]), K + (size_t)kn * HD_,
                  HD_, 32, HD_, T_, HD_);
      tdm_load_2d(lds_off(&sV[nxt][0][0]), V + kn,
                  32, HD_, T_, HD_, T_);
    }

    // S = Q @ K^T for 16q x 32k (two 16-key sub-tiles, d = 64)
    v8f s0 = {0.f,0.f,0.f,0.f,0.f,0.f,0.f,0.f};
    v8f s1 = s0;
    {
      v16bf k00 = frag_b(&sK[cur][ 0][0],      HD_, lane);
      v16bf k01 = frag_b(&sK[cur][ 0][0] + 32, HD_, lane);
      s0 = wmma_bf16(qf0, k00, s0);
      s0 = wmma_bf16(qf1, k01, s0);
      v16bf k10 = frag_b(&sK[cur][16][0],      HD_, lane);
      v16bf k11 = frag_b(&sK[cur][16][0] + 32, HD_, lane);
      s1 = wmma_bf16(qf0, k10, s1);
      s1 = wmma_bf16(qf1, k11, s1);
    }

    // scale + causal mask
#pragma unroll
    for (int j = 0; j < 8; ++j) {
      int qrow = q0 + rbase + j;
      s0[j] = ((kc + n)      <= qrow) ? s0[j] * scale : -1e30f;
      s1[j] = ((kc + 16 + n) <= qrow) ? s1[j] * scale : -1e30f;
    }

    // online softmax (row stats live across 16 lanes of each half-wave)
    float alpha[8];
#pragma unroll
    for (int j = 0; j < 8; ++j) {
      float mv = hmax16(fmaxf(s0[j], s1[j]));
      float mn = fmaxf(m[j], mv);
      alpha[j] = __expf(m[j] - mn);
      m[j] = mn;
      s0[j] = __expf(s0[j] - mn);
      s1[j] = __expf(s1[j] - mn);
      float rs = hsum16(s0[j] + s1[j]);
      l[j] = l[j] * alpha[j] + rs;
    }

    // rescale O, stage P into LDS (C-layout -> row-major [16][32])
#pragma unroll
    for (int j = 0; j < 8; ++j) {
      o0[j] *= alpha[j]; o1[j] *= alpha[j];
      o2[j] *= alpha[j]; o3[j] *= alpha[j];
      sP[wave][rbase + j][n]      = (bf16)s0[j];
      sP[wave][rbase + j][16 + n] = (bf16)s1[j];
    }
    // LDS is in-order per wave; enforce the DS RAW explicitly.
    asm volatile("s_wait_dscnt 0" ::: "memory");

    // O += P @ V
    v16bf pf  = frag_a(&sP[wave][0][0], 32, lane);
    v16bf v0f = frag_b(&sV[cur][ 0][0], 32, lane);
    v16bf v1f = frag_b(&sV[cur][16][0], 32, lane);
    v16bf v2f = frag_b(&sV[cur][32][0], 32, lane);
    v16bf v3f = frag_b(&sV[cur][48][0], 32, lane);
    o0 = wmma_bf16(pf, v0f, o0);
    o1 = wmma_bf16(pf, v1f, o1);
    o2 = wmma_bf16(pf, v2f, o2);
    o3 = wmma_bf16(pf, v3f, o3);

    if (wave == 0)
      __builtin_amdgcn_s_wait_tensorcnt(0);   // chunk c+1 landed
    __syncthreads();                          // publish to all waves
  }

  // normalize and write attention output as proj-GEMM A matrix:
  // Ao[(b*T + t)][h*64 + d] bf16
#pragma unroll
  for (int j = 0; j < 8; ++j) {
    float inv = 1.0f / l[j];
    size_t row = (size_t)b * T_ + q0 + rbase + j;
    bf16* dst = Ao + row * INNER_ + h * HD_;
    dst[ 0 + n] = (bf16)(o0[j] * inv);
    dst[16 + n] = (bf16)(o1[j] * inv);
    dst[32 + n] = (bf16)(o2[j] * inv);
    dst[48 + n] = (bf16)(o3[j] * inv);
  }
}

// ---------------------------------------------------------------------------

extern "C" void kernel_launch(void* const* d_in, const int* in_sizes, int n_in,
                              void* d_out, int out_size, void* d_ws, size_t ws_size,
                              hipStream_t stream) {
  (void)in_sizes; (void)n_in; (void)out_size; (void)ws_size;
  const float* x      = (const float*)d_in[0];
  const float* w_qkv  = (const float*)d_in[1];
  const float* w_proj = (const float*)d_in[2];
  float* out = (float*)d_out;

  // Workspace layout (~96 MB)
  char* w = (char*)d_ws;
  bf16* xb     = (bf16*)w;  w += (size_t)MROWS_ * DIM_ * 2;            //  8 MB
  bf16* wqkvT  = (bf16*)w;  w += (size_t)3 * INNER_ * DIM_ * 2;        //  6 MB
  bf16* wprojT = (bf16*)w;  w += (size_t)DIM_ * INNER_ * 2;            //  2 MB
  float* qkv   = (float*)w; w += (size_t)MROWS_ * 3 * INNER_ * 4;      // 48 MB
  bf16* Qh     = (bf16*)w;  w += (size_t)B_ * H_ * T_ * HD_ * 2;       //  8 MB
  bf16* Kh     = (bf16*)w;  w += (size_t)B_ * H_ * T_ * HD_ * 2;       //  8 MB
  bf16* Vt     = (bf16*)w;  w += (size_t)B_ * H_ * T_ * HD_ * 2;       //  8 MB
  bf16* Ao     = (bf16*)w;                                             //  8 MB

  // 1) fp32 -> bf16 conversions (weights transposed to [N][K])
  cvt_kernel<<<(MROWS_ * DIM_ / 4 + 255) / 256, 256, 0, stream>>>(
      x, xb, (long)MROWS_ * DIM_);
  cvt_t_kernel<<<((long)DIM_ * 3 * INNER_ + 255) / 256, 256, 0, stream>>>(
      w_qkv, wqkvT, DIM_, 3 * INNER_);
  cvt_t_kernel<<<((long)DIM_ * INNER_ + 255) / 256, 256, 0, stream>>>(
      w_proj, wprojT, DIM_, INNER_);

  // 2) qkv = x @ w_qkv   (4096 x 3072 x 1024, f32 out)
  gemm_bf16_kernel<<<dim3(3 * INNER_ / BN_, MROWS_ / BM_), 256, 0, stream>>>(
      xb, wqkvT, qkv, MROWS_, 3 * INNER_, DIM_);

  // 3) RoPE + scatter into head-major bf16 (V transposed)
  rope_kernel<<<(B_ * T_ * H_ * HD_) / 256, 256, 0, stream>>>(qkv, Qh, Kh, Vt);

  // 4) causal flash attention (TDM-staged K/V)
  attn_kernel<<<dim3(B_ * H_, T_ / 128), 256, 0, stream>>>(Qh, Kh, Vt, Ao);

  // 5) out = attn_out @ w_proj   (4096 x 1024 x 1024, f32 out)
  gemm_bf16_kernel<<<dim3(DIM_ / BN_, MROWS_ / BM_), 256, 0, stream>>>(
      Ao, wprojT, out, MROWS_, DIM_, INNER_);
}
